// MultiHeadAttention_1503238553720
// MI455X (gfx1250) — compile-verified, matrix operand order checked by reading, not run on hardware
//
#include <hip/hip_runtime.h>
#include <hip/hip_bf16.h>

#define D_MODEL 1024
#define HHEADS  16
#define DKDIM   64
#define S_LEN   2048
#define B_SZ    4
#define BS_TOK  (B_SZ * S_LEN)        // 8192 tokens
#define NEGBIG  (-1e9f)

typedef __attribute__((ext_vector_type(16))) __bf16 v16bf;
typedef __attribute__((ext_vector_type(8)))  __bf16 v8bf;
typedef __attribute__((ext_vector_type(8)))  float  v8f;
typedef __attribute__((ext_vector_type(4)))  unsigned int v4u;
typedef __attribute__((ext_vector_type(8)))  int v8i;
typedef __attribute__((ext_vector_type(4)))  int v4i;

#if defined(__has_builtin)
#  if __has_builtin(__builtin_amdgcn_tensor_load_to_lds)
#    define HAVE_TDM 1
#  endif
#endif
#ifndef HAVE_TDM
#  define HAVE_TDM 0
#endif

__device__ __forceinline__ v8f zero_v8f() {
  v8f z;
#pragma unroll
  for (int i = 0; i < 8; ++i) z[i] = 0.0f;
  return z;
}

// 16x32 bf16 A/B fragment for V_WMMA_*_16X16X32_BF16 (ISA 7.12.2):
// lane element e -> K = kbase + (e<8 ? e : e+8) + (lane>=16 ? 8 : 0)
__device__ __forceinline__ v16bf frag_from_bf16(const __bf16* rowptr, int kbase, int lane) {
  const int hi = (lane >> 4) & 1;
  const int c0 = kbase + hi * 8;
  const int c1 = c0 + 16;
  v8bf lo = *(const v8bf*)(rowptr + c0);   // 16B aligned
  v8bf hv = *(const v8bf*)(rowptr + c1);
  v16bf r;
#pragma unroll
  for (int i = 0; i < 8; ++i) { r[i] = lo[i]; r[i + 8] = hv[i]; }
  return r;
}

__device__ __forceinline__ v16bf frag_from_f32(const float* rowptr, int kbase, int lane) {
  const int hi = (lane >> 4) & 1;
  const int c0 = kbase + hi * 8;
  const int c1 = c0 + 16;
  v8f lo = *(const v8f*)(rowptr + c0);     // 32B aligned
  v8f hv = *(const v8f*)(rowptr + c1);
  v16bf r;
#pragma unroll
  for (int i = 0; i < 8; ++i) { r[i] = (__bf16)lo[i]; r[i + 8] = (__bf16)hv[i]; }
  return r;
}

__device__ __forceinline__ v8f wmma_bf16(v16bf a, v16bf b, v8f c) {
  return __builtin_amdgcn_wmma_f32_16x16x32_bf16(false, a, false, b, (short)0, c,
                                                 false, false);
}

#if HAVE_TDM
// TDM: DMA a 64-row x 32-col bf16 tile (row stride 1024 elems) to LDS.
// D# per CDNA5 ISA: group0 = {count=1, lds_addr,
// global_addr[56:0], type=2}; group1 = {data_size=1(2B), tensor dims 1024,
// tile_dim0=32, tile_dim1=64, tensor_dim0_stride=1024}. Tracked by TENSORcnt.
__device__ __forceinline__ void tdm_load_w_tile(unsigned lds_off, const __bf16* tile_gaddr) {
  const unsigned long long ga = (unsigned long long)(uintptr_t)tile_gaddr;
  v4u g0;
  g0[0] = 1u;                                                  // count=1
  g0[1] = lds_off;                                             // lds_addr
  g0[2] = (unsigned)ga;                                        // global_addr lo
  g0[3] = (unsigned)((ga >> 32) & 0x01ffffffu) | 0x80000000u;  // addr hi | type=2
  v8i g1;
  g1[0] = 0x00010000;                            // data_size=1 -> 2 bytes/elem
  g1[1] = (int)(1024u << 16);                    // tensor_dim0 = 1024 (lo16)
  g1[2] = (int)(1024u << 16);                    // dim0 hi=0 | tensor_dim1 lo = 1024
  g1[3] = (int)(32u << 16);                      // dim1 hi=0 | tile_dim0 = 32
  g1[4] = 64;                                    // tile_dim1 = 64, tile_dim2 = 0
  g1[5] = 1024;                                  // tensor_dim0_stride = 1024
  g1[6] = 0;
  g1[7] = 0;
  const v4i z4 = {0, 0, 0, 0};
#if defined(__clang_major__) && (__clang_major__ >= 23)
  const v8i z8 = {0, 0, 0, 0, 0, 0, 0, 0};
  __builtin_amdgcn_tensor_load_to_lds(g0, g1, z4, z4, z8, 0);
#else
  __builtin_amdgcn_tensor_load_to_lds(g0, g1, z4, z4, 0);
#endif
}
#endif  // HAVE_TDM

// ---------------------------------------------------------------------------
// Kernel 0: one-shot f32 -> bf16 conversion of the four weight matrices.
// grid = (512, 4), block = 256; 8 elems/thread.
// ---------------------------------------------------------------------------
__global__ __launch_bounds__(256) void convw_kernel(
    const float* __restrict__ wq, const float* __restrict__ wk,
    const float* __restrict__ wv, const float* __restrict__ wo,
    __bf16* __restrict__ dq, __bf16* __restrict__ dk,
    __bf16* __restrict__ dv, __bf16* __restrict__ dwo) {
  const float* src; __bf16* dst;
  switch (blockIdx.y) {
    case 0:  src = wq; dst = dq;  break;
    case 1:  src = wk; dst = dk;  break;
    case 2:  src = wv; dst = dv;  break;
    default: src = wo; dst = dwo; break;
  }
  const size_t i = ((size_t)blockIdx.x * 256 + threadIdx.x) * 8;
  v8f x = *(const v8f*)(src + i);
  v8bf y;
#pragma unroll
  for (int t = 0; t < 8; ++t) y[t] = (__bf16)x[t];
  *(v8bf*)(dst + i) = y;
}

// ---------------------------------------------------------------------------
// Kernel 1: fused Q/K/V projections.  Y = X @ W^T + b, bf16 out.
// grid = (BS/64, D/64, 3), block = 128.  Block = 4 row-tile waves sharing one
// 64-col W tile, TDM-staged into double-buffered LDS.
// Q,K stored [B,H,S,64]; V stored transposed [B,H,64,S].
// ---------------------------------------------------------------------------
__global__ __launch_bounds__(128) void proj_qkv_kernel(
    const float* __restrict__ qin, const float* __restrict__ kin,
    const float* __restrict__ vin,
    const __bf16* __restrict__ Wqb, const __bf16* __restrict__ Wkb,
    const __bf16* __restrict__ Wvb,
    const float* __restrict__ bq, const float* __restrict__ bk,
    const float* __restrict__ bv,
    __bf16* __restrict__ Qbf, __bf16* __restrict__ Kbf, __bf16* __restrict__ Vt) {
  __shared__ __bf16 wtile[2][64 * 32];   // 8 KB double buffer

  const int lane = threadIdx.x & 31;
  const int wid  = threadIdx.x >> 5;
  const int t0   = (blockIdx.x * 4 + wid) * 16;   // token tile base (per wave)
  const int c0   = blockIdx.y * 64;               // 64 output cols (per block)
  const int zsel = blockIdx.z;

  const float* X; const __bf16* Wb; const float* bias;
  if (zsel == 0)      { X = qin; Wb = Wqb; bias = bq; }
  else if (zsel == 1) { X = kin; Wb = Wkb; bias = bk; }
  else                { X = vin; Wb = Wvb; bias = bv; }

  const int nl = lane & 15;
  const int hi = (lane >> 4) & 1;
  const float* arowp = X + (size_t)(t0 + nl) * D_MODEL;

  v8f acc[4];
#pragma unroll
  for (int j = 0; j < 4; ++j) acc[j] = zero_v8f();

#if HAVE_TDM
  if (wid == 0)
    tdm_load_w_tile((unsigned)(uintptr_t)&wtile[0][0], Wb + (size_t)c0 * D_MODEL);
  for (int it = 0; it < D_MODEL / 32; ++it) {
    const int kk = it * 32, cur = it & 1;
    if (wid == 0) {
      if (it + 1 < D_MODEL / 32) {
        tdm_load_w_tile((unsigned)(uintptr_t)&wtile[cur ^ 1][0],
                        Wb + (size_t)c0 * D_MODEL + (kk + 32));
        __builtin_amdgcn_s_wait_tensorcnt(1);   // current tile done (in-order)
      } else {
        __builtin_amdgcn_s_wait_tensorcnt(0);
      }
    }
    __syncthreads();
    // batch all loads before the WMMA chain: 1 A frag (global f32) + 4 B frags
    // (LDS bf16) in flight together, one dscnt drain instead of four.
    const v16bf a = frag_from_f32(arowp, kk, lane);
    v16bf bfr[4];
#pragma unroll
    for (int j = 0; j < 4; ++j)
      bfr[j] = frag_from_bf16(&wtile[cur][(j * 16 + nl) * 32], 0, lane);
#pragma unroll
    for (int j = 0; j < 4; ++j) acc[j] = wmma_bf16(a, bfr[j], acc[j]);
    __syncthreads();   // everyone done with cur before it is restaged
  }
#else
  for (int kk = 0; kk < D_MODEL; kk += 32) {
    const v16bf a = frag_from_f32(arowp, kk, lane);
    v16bf bfr[4];
#pragma unroll
    for (int j = 0; j < 4; ++j)
      bfr[j] = frag_from_bf16(Wb + (size_t)(c0 + j * 16 + nl) * D_MODEL, kk, lane);
#pragma unroll
    for (int j = 0; j < 4; ++j) acc[j] = wmma_bf16(a, bfr[j], acc[j]);
  }
#endif

  // epilogue: + bias, cast bf16, scatter to head-major layouts
#pragma unroll
  for (int j = 0; j < 4; ++j) {
    const int c  = c0 + j * 16 + nl;
    const float bb = bias[c];
    const int h = c >> 6, d = c & 63;
#pragma unroll
    for (int r = 0; r < 8; ++r) {
      const int t = t0 + r + hi * 8;
      const int b = t >> 11;            // t / S_LEN
      const int s = t & (S_LEN - 1);
      const __bf16 vb = (__bf16)(acc[j][r] + bb);
      if (zsel == 2) {
        Vt[((size_t)(b * HHEADS + h) * DKDIM + d) * S_LEN + s] = vb;
      } else {
        __bf16* dst = (zsel == 0) ? Qbf : Kbf;
        dst[((size_t)(b * HHEADS + h) * S_LEN + s) * DKDIM + d] = vb;
      }
    }
  }
}

// ---------------------------------------------------------------------------
// Kernel 2: flash attention per head.  grid=(S/64, H, B), block=128.
// Each wave owns a 16-row query tile; streams keys/values in 32-key chunks.
// ---------------------------------------------------------------------------
__global__ __launch_bounds__(128) void attn_kernel(
    const __bf16* __restrict__ Qbf, const __bf16* __restrict__ Kbf,
    const __bf16* __restrict__ Vt, const int* __restrict__ mask,
    __bf16* __restrict__ Obuf) {
  __shared__ float pshare[4][16 * 32];   // per-wave P scratch (no cross-wave use)

  const int lane = threadIdx.x & 31;
  const int wid  = threadIdx.x >> 5;
  const int h = blockIdx.y, b = blockIdx.z;
  const int q0 = (blockIdx.x * 4 + wid) * 16;

  const size_t bh = (size_t)(b * HHEADS + h);
  const __bf16* Qb = Qbf + bh * S_LEN * DKDIM;
  const __bf16* Kb = Kbf + bh * S_LEN * DKDIM;
  const __bf16* Vb = Vt  + bh * DKDIM * S_LEN;
  const int* mrow = mask + (size_t)b * S_LEN * S_LEN;

  const int nl = lane & 15;
  const int hi = (lane >> 4) & 1;

  const __bf16* qrow = Qb + (size_t)(q0 + nl) * DKDIM;
  const v16bf qa0 = frag_from_bf16(qrow, 0, lane);
  const v16bf qa1 = frag_from_bf16(qrow, 32, lane);

  v8f acc[4];
#pragma unroll
  for (int j = 0; j < 4; ++j) acc[j] = zero_v8f();
  float rmax[8], rsum[8];
#pragma unroll
  for (int r = 0; r < 8; ++r) { rmax[r] = -1e30f; rsum[r] = 0.0f; }

  float* pw = &pshare[wid][0];

  for (int kb = 0; kb < S_LEN; kb += 32) {
    if (kb + 32 < S_LEN) {
      __builtin_prefetch(Kb + (size_t)(kb + 32 + nl) * DKDIM, 0, 1);
      __builtin_prefetch(Vb + (size_t)nl * S_LEN + kb + 32, 0, 1);
    }
    // ---- scores: S = Q(16x64) @ K^T(64x32); batch all 4 K frags first ----
    const __bf16* krow0 = Kb + (size_t)(kb + nl) * DKDIM;
    const __bf16* krow1 = Kb + (size_t)(kb + 16 + nl) * DKDIM;
    v16bf kf[4];
    kf[0] = frag_from_bf16(krow0, 0, lane);
    kf[1] = frag_from_bf16(krow0, 32, lane);
    kf[2] = frag_from_bf16(krow1, 0, lane);
    kf[3] = frag_from_bf16(krow1, 32, lane);
    v8f s1 = zero_v8f(), s2 = zero_v8f();
    s1 = wmma_bf16(qa0, kf[0], s1);
    s1 = wmma_bf16(qa1, kf[1], s1);
    s2 = wmma_bf16(qa0, kf[2], s2);
    s2 = wmma_bf16(qa1, kf[3], s2);

    // ---- scale + mask in C/D layout: element (m,n): m=r+8*hi, n=lane&15 ----
#pragma unroll
    for (int r = 0; r < 8; ++r) {
      const int m = q0 + r + hi * 8;
      float x1 = s1[r] * 0.125f;                         // 1/sqrt(64)
      float x2 = s2[r] * 0.125f;
      if (mrow[(size_t)m * S_LEN + kb + nl] == 0)        x1 = NEGBIG;
      if (mrow[(size_t)m * S_LEN + kb + 16 + nl] == 0)   x2 = NEGBIG;
      s1[r] = x1; s2[r] = x2;
    }

    // ---- online softmax: row stats live in matching (r, half) layout ----
#pragma unroll
    for (int r = 0; r < 8; ++r) {
      float tm = fmaxf(s1[r], s2[r]);
      tm = fmaxf(tm, __shfl_xor(tm, 1, 32));
      tm = fmaxf(tm, __shfl_xor(tm, 2, 32));
      tm = fmaxf(tm, __shfl_xor(tm, 4, 32));
      tm = fmaxf(tm, __shfl_xor(tm, 8, 32));
      const float nm = fmaxf(rmax[r], tm);
      const float sf = __expf(rmax[r] - nm);
      const float e1 = __expf(s1[r] - nm);
      const float e2 = __expf(s2[r] - nm);
      s1[r] = e1; s2[r] = e2;
      float ts = e1 + e2;
      ts += __shfl_xor(ts, 1, 32);
      ts += __shfl_xor(ts, 2, 32);
      ts += __shfl_xor(ts, 4, 32);
      ts += __shfl_xor(ts, 8, 32);
      rsum[r] = rsum[r] * sf + ts;
      rmax[r] = nm;
#pragma unroll
      for (int j = 0; j < 4; ++j) acc[j][r] *= sf;
    }

    // ---- D-layout f32 -> A-layout bf16 via per-wave LDS slice ----
#pragma unroll
    for (int r = 0; r < 8; ++r) {
      const int ml = r + hi * 8;
      pw[ml * 32 + nl]      = s1[r];
      pw[ml * 32 + 16 + nl] = s2[r];
    }
    __builtin_amdgcn_wave_barrier();     // keep compiler from moving DS ops
    const float* pr = pw + nl * 32;      // A row m = lane&15
    v16bf pf;
    const int cp = hi * 8;
#pragma unroll
    for (int i = 0; i < 8; ++i) {
      pf[i]     = (__bf16)pr[cp + i];
      pf[i + 8] = (__bf16)pr[16 + cp + i];
    }
    __builtin_amdgcn_wave_barrier();

    // ---- O += P(16x32) @ V(32x16); batch all 4 V frags before WMMA chain ----
    v16bf vf[4];
#pragma unroll
    for (int j = 0; j < 4; ++j)
      vf[j] = frag_from_bf16(Vb + (size_t)(j * 16 + nl) * S_LEN, kb, lane);
#pragma unroll
    for (int j = 0; j < 4; ++j) acc[j] = wmma_bf16(pf, vf[j], acc[j]);
  }

  // ---- normalize and store to Obuf [B,S,D] bf16 ----
#pragma unroll
  for (int j = 0; j < 4; ++j) {
    const int d = h * DKDIM + j * 16 + nl;
#pragma unroll
    for (int r = 0; r < 8; ++r) {
      const int m = q0 + r + hi * 8;
      const float val = acc[j][r] / rsum[r];
      Obuf[((size_t)b * S_LEN + m) * D_MODEL + d] = (__bf16)val;
    }
  }
}

// ---------------------------------------------------------------------------
// Kernel 3: output projection  out = Obuf(bf16) @ wo^T + bo  (f32 out)
// grid = (BS/64, D/64), block = 128; TDM-staged W tiles as in kernel 1.
// ---------------------------------------------------------------------------
__global__ __launch_bounds__(128) void out_proj_kernel(
    const __bf16* __restrict__ A, const __bf16* __restrict__ Wob,
    const float* __restrict__ bo, float* __restrict__ out) {
  __shared__ __bf16 wtile[2][64 * 32];

  const int lane = threadIdx.x & 31;
  const int wid  = threadIdx.x >> 5;
  const int t0 = (blockIdx.x * 4 + wid) * 16;
  const int c0 = blockIdx.y * 64;
  const int nl = lane & 15;
  const int hi = (lane >> 4) & 1;

  const __bf16* arow = A + (size_t)(t0 + nl) * D_MODEL;
  v8f acc[4];
#pragma unroll
  for (int j = 0; j < 4; ++j) acc[j] = zero_v8f();

#if HAVE_TDM
  if (wid == 0)
    tdm_load_w_tile((unsigned)(uintptr_t)&wtile[0][0], Wob + (size_t)c0 * D_MODEL);
  for (int it = 0; it < D_MODEL / 32; ++it) {
    const int kk = it * 32, cur = it & 1;
    if (wid == 0) {
      if (it + 1 < D_MODEL / 32) {
        tdm_load_w_tile((unsigned)(uintptr_t)&wtile[cur ^ 1][0],
                        Wob + (size_t)c0 * D_MODEL + (kk + 32));
        __builtin_amdgcn_s_wait_tensorcnt(1);
      } else {
        __builtin_amdgcn_s_wait_tensorcnt(0);
      }
    }
    __syncthreads();
    const v16bf a = frag_from_bf16(arow, kk, lane);
    v16bf bfr[4];
#pragma unroll
    for (int j = 0; j < 4; ++j)
      bfr[j] = frag_from_bf16(&wtile[cur][(j * 16 + nl) * 32], 0, lane);
#pragma unroll
    for (int j = 0; j < 4; ++j) acc[j] = wmma_bf16(a, bfr[j], acc[j]);
    __syncthreads();
  }
#else
  for (int kk = 0; kk < D_MODEL; kk += 32) {
    const v16bf a = frag_from_bf16(arow, kk, lane);
    v16bf bfr[4];
#pragma unroll
    for (int j = 0; j < 4; ++j)
      bfr[j] = frag_from_bf16(Wob + (size_t)(c0 + j * 16 + nl) * D_MODEL, kk, lane);
#pragma unroll
    for (int j = 0; j < 4; ++j) acc[j] = wmma_bf16(a, bfr[j], acc[j]);
  }
#endif

#pragma unroll
  for (int j = 0; j < 4; ++j) {
    const int c = c0 + j * 16 + nl;
    const float bb = bo[c];
#pragma unroll
    for (int r = 0; r < 8; ++r) {
      const int t = t0 + r + hi * 8;
      out[(size_t)t * D_MODEL + c] = acc[j][r] + bb;
    }
  }
}

// ---------------------------------------------------------------------------
extern "C" void kernel_launch(void* const* d_in, const int* in_sizes, int n_in,
                              void* d_out, int out_size, void* d_ws, size_t ws_size,
                              hipStream_t stream) {
  (void)in_sizes; (void)n_in; (void)out_size; (void)ws_size;
  const float* q  = (const float*)d_in[0];
  const float* k  = (const float*)d_in[1];
  const float* v  = (const float*)d_in[2];
  const int*  msk = (const int*)d_in[3];
  const float* wq = (const float*)d_in[4];
  const float* bq = (const float*)d_in[5];
  const float* wk = (const float*)d_in[6];
  const float* bk = (const float*)d_in[7];
  const float* wv = (const float*)d_in[8];
  const float* bv = (const float*)d_in[9];
  const float* wo = (const float*)d_in[10];
  const float* bo = (const float*)d_in[11];
  float* out = (float*)d_out;

  // workspace: 4 x 16MB bf16 activation buffers + 4 x 2MB bf16 weights = 72MB
  const size_t ELEMS = (size_t)B_SZ * HHEADS * S_LEN * DKDIM;  // 8388608
  const size_t WELEMS = (size_t)D_MODEL * D_MODEL;             // 1048576
  __bf16* Qbf  = (__bf16*)d_ws;
  __bf16* Kbf  = Qbf + ELEMS;
  __bf16* Vt   = Kbf + ELEMS;
  __bf16* Obuf = Vt + ELEMS;
  __bf16* Wqb  = Obuf + ELEMS;
  __bf16* Wkb  = Wqb + WELEMS;
  __bf16* Wvb  = Wkb + WELEMS;
  __bf16* Wob  = Wvb + WELEMS;

  convw_kernel<<<dim3(WELEMS / (256 * 8), 4, 1), dim3(256, 1, 1), 0, stream>>>(
      wq, wk, wv, wo, Wqb, Wkb, Wvb, Wob);

  dim3 blk(128, 1, 1);
  proj_qkv_kernel<<<dim3(BS_TOK / 64, D_MODEL / 64, 3), blk, 0, stream>>>(
      q, k, v, Wqb, Wkb, Wvb, bq, bk, bv, Qbf, Kbf, Vt);
  attn_kernel<<<dim3(S_LEN / 64, HHEADS, B_SZ), blk, 0, stream>>>(
      Qbf, Kbf, Vt, msk, Obuf);
  out_proj_kernel<<<dim3(BS_TOK / 64, D_MODEL / 64, 1), blk, 0, stream>>>(
      Obuf, Wob, bo, out);
}
